// CHopfield_40020505264706
// MI455X (gfx1250) — matrix operand order repaired and checked by reading, not compiled, not run on hardware
//
#include <hip/hip_runtime.h>

// Hopfield iterative update on MI455X (gfx1250).
// bf16 WMMA (v_wmma_f32_16x16x32_bf16) for both GEMMs, fp32 online softmax,
// flash-style fusion so the 4096x8192 score matrix never leaves LDS.
// M=32 row tile / 512 threads: halves L2 pattern traffic vs M=16 (4 GB/iter),
// per-wave register pressure unchanged (8 f32 C-tiles = 64 acc VGPRs).
// Workspace layout (40 MB total):
//   [0,16MB)   patterns bf16, row-major [8192][1024]   (GEMM1 B operand)
//   [16,32MB)  patterns bf16, transposed [1024][8192]  (GEMM2 B operand)
//   [32,40MB)  x bf16 [4096][1024]                     (GEMM1 A operand)

typedef __bf16 bf16;
typedef __attribute__((ext_vector_type(16))) __bf16        v16bf;
typedef __attribute__((ext_vector_type(8)))  float         v8f;
typedef __attribute__((ext_vector_type(4)))  unsigned int  u32x4;

#define B_ROWS 4096
#define NPAT   8192
#define NNEU   1024
#define TEMP   1.0f

#define TILE_M 32          // rows of x per block
#define TILE_P 256         // patterns per streaming tile (16 waves x 16)
#define NTHREADS 512
// LDS row strides padded by 16 bytes so 16 lanes with consecutive rows hit
// disjoint bank groups on 128-bit ds loads.
#define XLDS_STRIDE 1032   // 1024 bf16 + 8 pad
#define SLDS_STRIDE 260    // 256 f32  + 4 pad
#define WLDS_STRIDE 264    // 256 bf16 + 8 pad

union V16 { v16bf v; u32x4 q[2]; };

__global__ __launch_bounds__(256)
void cvt_patterns(const float* __restrict__ pat,
                  bf16* __restrict__ pat_bf,   // [NPAT][NNEU]
                  bf16* __restrict__ pt_bf)    // [NNEU][NPAT]
{
  size_t i = (size_t)blockIdx.x * 256 + threadIdx.x;   // exact multiple
  float v = pat[i];
  bf16  b = (bf16)v;
  pat_bf[i] = b;
  size_t p = i >> 10;          // i / NNEU
  size_t n = i & (NNEU - 1);   // i % NNEU
  pt_bf[n * (size_t)NPAT + p] = b;
}

__global__ __launch_bounds__(256)
void cvt_x(const float* __restrict__ src, bf16* __restrict__ dst)
{
  size_t i = (size_t)blockIdx.x * 256 + threadIdx.x;
  dst[i] = (bf16)src[i];
}

__global__ __launch_bounds__(NTHREADS)
void hopfield_step(const bf16* __restrict__ x_bf,    // [B][NNEU]
                   const bf16* __restrict__ pat_bf,  // [NPAT][NNEU]
                   const bf16* __restrict__ pt_bf,   // [NNEU][NPAT]
                   float* __restrict__ out)          // [B][NNEU]
{
  __shared__ __align__(16) bf16  x_lds[TILE_M * XLDS_STRIDE];   // 66.0 KB
  __shared__ __align__(16) float S_lds[TILE_M * SLDS_STRIDE];   // 33.3 KB
  __shared__ __align__(16) bf16  W_lds[TILE_M * WLDS_STRIDE];   // 16.9 KB
  __shared__ float scale_lds[TILE_M];
  __shared__ float linv_lds[TILE_M];

  const int tid  = threadIdx.x;
  const int lane = tid & 31;
  const int wv   = tid >> 5;      // wave 0..15
  const int mrow = lane & 15;     // WMMA A-row / B-col index for this lane
  const int half = lane >> 4;     // lane half (selects K sub-chunk)
  const int b0   = blockIdx.x * TILE_M;

  // ---- stage x tile (32 x 1024 bf16) into LDS, padded rows ----
  #pragma unroll
  for (int c = 0; c < 8; ++c) {
    int e   = (tid + NTHREADS * c) * 8;   // bf16 element index, 8 per 16B chunk
    int r   = e >> 10;
    int col = e & 1023;
    *(u32x4*)&x_lds[r * XLDS_STRIDE + col] =
        *(const u32x4*)&x_bf[(size_t)(b0 + r) * NNEU + col];
  }
  __syncthreads();

  v8f oacc[2][4] = {};             // [M-tile][N-tile] 16x16 f32 per wave
  float m_run = -3.0e38f;
  float l_run = 0.0f;
  const int srow = tid >> 4;       // softmax row owned by this thread (0..31)
  const int scol = tid & 15;

  for (int p0 = 0; p0 < NPAT; p0 += TILE_P) {
    // ------ GEMM1: S_tile = x_tile @ patterns^T (B regs reused over 2 M) ---
    {
      v8f sacc0 = {}, sacc1 = {};
      const bf16* bsrc  = pat_bf + (size_t)(p0 + 16 * wv + mrow) * NNEU + 8 * half;
      const bf16* arow0 = &x_lds[mrow * XLDS_STRIDE + 8 * half];
      const bf16* arow1 = &x_lds[(16 + mrow) * XLDS_STRIDE + 8 * half];
      #pragma unroll 4
      for (int kk = 0; kk < NNEU; kk += 32) {
        V16 a0, a1, b;
        b.q[0]  = *(const u32x4*)(bsrc + kk);        // B: K=kk+8h .. +7, kk+16+8h .. +7
        b.q[1]  = *(const u32x4*)(bsrc + kk + 16);
        a0.q[0] = *(const u32x4*)(arow0 + kk);       // A: K=kk+8h .. +7
        a0.q[1] = *(const u32x4*)(arow0 + kk + 16);  // A: K=kk+16+8h .. +7
        a1.q[0] = *(const u32x4*)(arow1 + kk);
        a1.q[1] = *(const u32x4*)(arow1 + kk + 16);
        sacc0 = __builtin_amdgcn_wmma_f32_16x16x32_bf16(
            false, a0.v, false, b.v, (short)0, sacc0, false, false);
        sacc1 = __builtin_amdgcn_wmma_f32_16x16x32_bf16(
            false, a1.v, false, b.v, (short)0, sacc1, false, false);
      }
      #pragma unroll
      for (int r = 0; r < 8; ++r) {
        S_lds[(r + 8 * half) * SLDS_STRIDE + 16 * wv + mrow]        = sacc0[r];
        S_lds[(16 + r + 8 * half) * SLDS_STRIDE + 16 * wv + mrow]   = sacc1[r];
      }
    }
    __syncthreads();

    // ---------------- online softmax over this 256-pattern tile -----------
    {
      float s[16], mloc = -3.0e38f;
      #pragma unroll
      for (int j = 0; j < 16; ++j) {
        s[j] = TEMP * S_lds[srow * SLDS_STRIDE + scol + 16 * j];
        mloc = fmaxf(mloc, s[j]);
      }
      #pragma unroll
      for (int d = 1; d < 16; d <<= 1)
        mloc = fmaxf(mloc, __shfl_xor(mloc, d, 32));
      float mnew = fmaxf(m_run, mloc);
      float corr = __expf(m_run - mnew);
      float ssum = 0.0f;
      #pragma unroll
      for (int j = 0; j < 16; ++j) {
        float w = __expf(s[j] - mnew);
        ssum += w;
        W_lds[srow * WLDS_STRIDE + scol + 16 * j] = (bf16)w;
      }
      #pragma unroll
      for (int d = 1; d < 16; d <<= 1)
        ssum += __shfl_xor(ssum, d, 32);
      l_run = l_run * corr + ssum;
      m_run = mnew;
      if (scol == 0) scale_lds[srow] = corr;
    }
    __syncthreads();

    // ------- rescale O, GEMM2: O += exp(S) @ P_tile (B reused over 2 M) ----
    {
      #pragma unroll
      for (int mt = 0; mt < 2; ++mt) {
        #pragma unroll
        for (int r = 0; r < 8; ++r) {
          float corr = scale_lds[mt * 16 + r + 8 * half];
          #pragma unroll
          for (int t = 0; t < 4; ++t) oacc[mt][t][r] *= corr;
        }
      }
      const bf16* wrow0 = &W_lds[mrow * WLDS_STRIDE + 8 * half];
      const bf16* wrow1 = &W_lds[(16 + mrow) * WLDS_STRIDE + 8 * half];
      const bf16* bbase = pt_bf + (size_t)(64 * wv + mrow) * NPAT + p0 + 8 * half;
      #pragma unroll
      for (int kk = 0; kk < TILE_P; kk += 32) {
        V16 a0, a1;                                  // A reused across 4 N-tiles
        a0.q[0] = *(const u32x4*)(wrow0 + kk);
        a0.q[1] = *(const u32x4*)(wrow0 + kk + 16);
        a1.q[0] = *(const u32x4*)(wrow1 + kk);
        a1.q[1] = *(const u32x4*)(wrow1 + kk + 16);
        #pragma unroll
        for (int t = 0; t < 4; ++t) {
          V16 b;                                     // B reused across 2 M-tiles
          const bf16* bp = bbase + (size_t)(16 * t) * NPAT + kk;
          b.q[0] = *(const u32x4*)(bp);
          b.q[1] = *(const u32x4*)(bp + 16);
          oacc[0][t] = __builtin_amdgcn_wmma_f32_16x16x32_bf16(
              false, a0.v, false, b.v, (short)0, oacc[0][t], false, false);
          oacc[1][t] = __builtin_amdgcn_wmma_f32_16x16x32_bf16(
              false, a1.v, false, b.v, (short)0, oacc[1][t], false, false);
        }
      }
    }
    __syncthreads();
  }

  if (scol == 0) linv_lds[srow] = 1.0f / l_run;
  __syncthreads();

  #pragma unroll
  for (int mt = 0; mt < 2; ++mt) {
    float li[8];
    #pragma unroll
    for (int r = 0; r < 8; ++r) li[r] = linv_lds[mt * 16 + r + 8 * half];
    #pragma unroll
    for (int t = 0; t < 4; ++t) {
      int n = 64 * wv + 16 * t + mrow;
      #pragma unroll
      for (int r = 0; r < 8; ++r)
        out[(size_t)(b0 + mt * 16 + r + 8 * half) * NNEU + n] = oacc[mt][t][r] * li[r];
    }
  }
}

extern "C" void kernel_launch(void* const* d_in, const int* in_sizes, int n_in,
                              void* d_out, int out_size, void* d_ws, size_t ws_size,
                              hipStream_t stream) {
  (void)in_sizes; (void)n_in; (void)out_size; (void)ws_size;
  const float* x_f32   = (const float*)d_in[0];   // [4096][1024]
  const float* pat_f32 = (const float*)d_in[1];   // [8192][1024]
  float* out = (float*)d_out;                     // [4096][1024]

  char* ws = (char*)d_ws;
  bf16* pat_bf = (bf16*)ws;                              // 16 MB
  bf16* pt_bf  = (bf16*)(ws + ((size_t)16 << 20));       // 16 MB
  bf16* x_bf   = (bf16*)(ws + ((size_t)32 << 20));       //  8 MB

  // One-time bf16 conversion + transpose of patterns (L2-resident afterwards).
  cvt_patterns<<<(NPAT * NNEU) / 256, 256, 0, stream>>>(pat_f32, pat_bf, pt_bf);

  for (int it = 0; it < 3; ++it) {
    const float* src = (it == 0) ? x_f32 : out;
    cvt_x<<<(B_ROWS * NNEU) / 256, 256, 0, stream>>>(src, x_bf);
    hopfield_step<<<B_ROWS / TILE_M, NTHREADS, 0, stream>>>(x_bf, pat_bf, pt_bf, out);
  }
}
